// GLMultiHeadAttention_23184233464577
// MI455X (gfx1250) — matrix operand order repaired and checked by reading, not compiled, not run on hardware
//
#include <hip/hip_runtime.h>
#include <hip/hip_bf16.h>

typedef __attribute__((ext_vector_type(16))) _Float16 v16h;
typedef __attribute__((ext_vector_type(8)))  float    v8f;
typedef __attribute__((ext_vector_type(8)))  _Float16 h8v;
typedef __attribute__((ext_vector_type(4)))  float    f4v;

constexpr int NB  = 2;
constexpr int NSG = 128;
constexpr int NSL = 2048;
constexpr int NS  = 2176;   // NSL + NSG
constexpr int ND  = 768;
constexpr int NDH = 384;
constexpr int NH  = 8;
constexpr int NHD = 48;
constexpr float INV_SCALE = 0.14433756729740643f; // 1/sqrt(48)

// ---------------- WMMA helpers ----------------

__device__ __forceinline__ v8f wmma16(v16h a, v16h b, v8f c) {
    return __builtin_amdgcn_wmma_f32_16x16x32_f16(
        false, a, false, b, (short)0, c, false, false);
}

__device__ __forceinline__ v16h combine16(h8v x, h8v y) {
    v16h a;
#pragma unroll
    for (int i = 0; i < 8; ++i) { a[i] = x[i]; a[8 + i] = y[i]; }
    return a;
}

__device__ __forceinline__ h8v cvt8(f4v a, f4v b) {
    h8v r;
#pragma unroll
    for (int i = 0; i < 4; ++i) { r[i] = (_Float16)a[i]; r[4 + i] = (_Float16)b[i]; }
    return r;
}

// A fragment (16x32, MxK) from row-major f32 (row stride ld). 16B-aligned rows/k0.
__device__ __forceinline__ v16h loadA_f32(const float* __restrict__ src, int ld,
                                          int lane, int k0) {
    int m = lane & 15, hi = lane >> 4;
    const float* p = src + (size_t)m * ld + k0 + (hi << 4);
    f4v x0 = *(const f4v*)(p);
    f4v x1 = *(const f4v*)(p + 4);
    f4v y0 = *(const f4v*)(p + 8);
    f4v y1 = *(const f4v*)(p + 12);
    return combine16(cvt8(x0, x1), cvt8(y0, y1));
}

// A fragment (16x32) from row-major f16, no guard.
__device__ __forceinline__ v16h loadA_f16(const _Float16* __restrict__ src, int ld,
                                          int lane, int k0) {
    int m = lane & 15, hi = lane >> 4;
    const _Float16* p = src + (size_t)m * ld + k0 + (hi << 4);
    return combine16(*(const h8v*)(p), *(const h8v*)(p + 8));
}

// A fragment (16x32) from row-major f16 with chunk-granular K guard (kmax % 8 == 0).
__device__ __forceinline__ v16h loadA_f16g(const _Float16* __restrict__ src, int ld,
                                           int lane, int k0, int kmax) {
    int m = lane & 15, hi = lane >> 4;
    int s0 = k0 + (hi << 4), s1 = s0 + 8;
    const _Float16* p = src + (size_t)m * ld;
    h8v z = {};
    h8v x = (s0 < kmax) ? *(const h8v*)(p + s0) : z;
    h8v y = (s1 < kmax) ? *(const h8v*)(p + s1) : z;
    return combine16(x, y);
}

// B fragment (32x16, KxN) from N-major rows (W[n][k], row stride ld), f32, no guards.
__device__ __forceinline__ v16h loadB_f32(const float* __restrict__ src, int ld,
                                          int lane, int n0, int k0) {
    int n = lane & 15, hi = lane >> 4;
    const float* p = src + (size_t)(n0 + n) * ld + k0 + (hi << 4);
    f4v x0 = *(const f4v*)(p);
    f4v x1 = *(const f4v*)(p + 4);
    f4v x2 = *(const f4v*)(p + 8);
    f4v x3 = *(const f4v*)(p + 12);
    return combine16(cvt8(x0, x1), cvt8(x2, x3));
}

// B fragment f32 with row clamp and 16-block K guard (for rel tables; kmax % 16 == 0).
__device__ __forceinline__ v16h loadB_f32cg(const float* __restrict__ src, int ld,
                                            int lane, int n0, int nmax, int k0, int kmax) {
    int n = lane & 15, hi = lane >> 4;
    int row = n0 + n; if (row >= nmax) row = nmax - 1;
    int ks = k0 + (hi << 4);
    const float* p = src + (size_t)row * ld + ks;
    h8v z = {};
    if (ks < kmax) {
        f4v x0 = *(const f4v*)(p);
        f4v x1 = *(const f4v*)(p + 4);
        f4v x2 = *(const f4v*)(p + 8);
        f4v x3 = *(const f4v*)(p + 12);
        return combine16(cvt8(x0, x1), cvt8(x2, x3));
    }
    return combine16(z, z);
}

// B fragment f16 with row clamp and 16-block K guard (K^T from row-major keys).
__device__ __forceinline__ v16h loadB_f16cg(const _Float16* __restrict__ src, int ld,
                                            int lane, int n0, int nmax, int k0, int kmax) {
    int n = lane & 15, hi = lane >> 4;
    int row = n0 + n; if (row >= nmax) row = nmax - 1;
    int ks = k0 + (hi << 4);
    const _Float16* p = src + (size_t)row * ld + ks;
    h8v z = {};
    bool ok = ks < kmax;
    h8v x = ok ? *(const h8v*)(p) : z;
    h8v y = ok ? *(const h8v*)(p + 8) : z;
    return combine16(x, y);
}

// B fragment from V stored transposed (b,h,hd,seqk): rows = hd, K = key.
// 16-key blocks never straddle seqk; fully-OOB blocks are clamped back (their P==0).
__device__ __forceinline__ v16h loadB_vt(const _Float16* __restrict__ src, int ld,
                                         int lane, int hd0, int kb, int seqk) {
    int n = lane & 15, hi = lane >> 4;
    int ks = kb + (hi << 4);
    if (ks + 16 > seqk) ks = seqk - 16;
    const _Float16* p = src + (size_t)(hd0 + n) * ld + ks;
    return combine16(*(const h8v*)(p), *(const h8v*)(p + 8));
}

// ---------------- Stage 1: input projections, split-heads, f16 ----------------
// out = X[:, s_off:s_off+seq] @ W^T   -> (b,h,seq,hd) f16 (vt=0) or (b,h,hd,seq) (vt=1)
__global__ __launch_bounds__(32) void proj_kernel(const float* __restrict__ X,
                                                  const float* __restrict__ W,
                                                  _Float16* __restrict__ out,
                                                  int seq, int s_off, int vt) {
    int lane = threadIdx.x;
    int mt = blockIdx.x, nt = blockIdx.y;
    int rowbase = mt * 16;
    int b = rowbase / seq, q = rowbase % seq;
    int h = nt / 3, hd0 = (nt % 3) * 16;
    const float* Arow = X + ((size_t)b * NS + s_off + q) * ND;
    const float* Wrow = W + (size_t)nt * 16 * ND;
    v8f c = {};
    for (int k0 = 0; k0 < ND; k0 += 32) {
        c = wmma16(loadA_f32(Arow, ND, lane, k0), loadB_f32(Wrow, ND, lane, 0, k0), c);
    }
    int n = lane & 15, hi = lane >> 4;
    if (vt) {
        h8v o;
#pragma unroll
        for (int v = 0; v < 8; ++v) o[v] = (_Float16)c[v];
        _Float16* dst = out + (((size_t)b * NH + h) * NHD + hd0 + n) * seq + q + (hi << 3);
        *(h8v*)dst = o;
    } else {
        _Float16* dst = out + (((size_t)b * NH + h) * seq + q + (hi << 3)) * NHD + hd0 + n;
#pragma unroll
        for (int v = 0; v < 8; ++v) dst[(size_t)v * NHD] = (_Float16)c[v];
    }
}

// ---------------- Stage 2: q_rel = q @ rel^T (per head slice), f32 out padded to 32 ----
__global__ __launch_bounds__(32) void qrel_kernel(const _Float16* __restrict__ qws,
                                                  const float* __restrict__ rel,
                                                  float* __restrict__ qrel,
                                                  int seq, int R) {
    int lane = threadIdx.x;
    int mt = blockIdx.x, nt = blockIdx.y; // nt in 0..1
    int rowbase = mt * 16;
    int bh = rowbase / seq, q = rowbase % seq;
    int h = bh % NH;
    const _Float16* Arow = qws + ((size_t)bh * seq + q) * NHD;
    const float* relh = rel + (size_t)h * NHD; // row stride NDH, head slice
    v8f c = {};
    c = wmma16(loadA_f16g(Arow, NHD, lane, 0, NHD),
               loadB_f32cg(relh, NDH, lane, nt * 16, R, 0, NHD), c);
    c = wmma16(loadA_f16g(Arow, NHD, lane, 32, NHD),
               loadB_f32cg(relh, NDH, lane, nt * 16, R, 32, NHD), c);
    int n = lane & 15, hi = lane >> 4;
    float* dst = qrel + ((size_t)bh * seq + (q + (hi << 3))) * 32 + nt * 16 + n;
#pragma unroll
    for (int v = 0; v < 8; ++v) dst[(size_t)v * 32] = c[v];
}

// ---------------- Stage 3: flash attention (one wave = one (b,h,16-query tile)) -------
// mode: 0=l2l (sliding, md=12, radius 84), 1=l2g (seg, md=4), 2=g2g (md=8), 3=g2l (seg, md=4)
__global__ __launch_bounds__(32) void flash_kernel(const _Float16* __restrict__ qws,
                                                   const _Float16* __restrict__ kws,
                                                   const _Float16* __restrict__ vtws,
                                                   const float* __restrict__ qrel,
                                                   const unsigned char* __restrict__ pad,
                                                   const int* __restrict__ seg,
                                                   _Float16* __restrict__ ctx,
                                                   int seqq, int seqk, int mode, int pad_off) {
    __shared__ _Float16 plds[16 * 32];
    int lane = threadIdx.x;
    int nl = lane & 15, hi = lane >> 4;
    int qt = blockIdx.x;
    int bh = blockIdx.y;
    int b = bh / NH, h = bh % NH;
    int q0 = qt * 16;

    const _Float16* qbase = qws + ((size_t)bh * seqq + q0) * NHD;
    v16h qa0 = loadA_f16g(qbase, NHD, lane, 0, NHD);
    v16h qa1 = loadA_f16g(qbase, NHD, lane, 32, NHD);
    const _Float16* kbase_p = kws + (size_t)bh * seqk * NHD;
    const _Float16* vbase_p = vtws + (size_t)bh * NHD * seqk;
    const float* qrl = qrel + (size_t)bh * seqq * 32;

    float Mx[8], Ls[8];
    v8f O0 = {}, O1 = {}, O2 = {};
#pragma unroll
    for (int v = 0; v < 8; ++v) { Mx[v] = -1e30f; Ls[v] = 0.0f; }

    int segq[8] = {0, 0, 0, 0, 0, 0, 0, 0};
    if (mode == 1) {
#pragma unroll
        for (int v = 0; v < 8; ++v) segq[v] = seg[(size_t)b * NSL + q0 + hi * 8 + v];
    }

    int nkt = seqk >> 4;
    int kt0 = 0, kt1 = nkt - 1;
    if (mode == 0) {
        kt0 = qt - 6; if (kt0 < 0) kt0 = 0;
        kt1 = qt + 6; if (kt1 > nkt - 1) kt1 = nkt - 1;
    }

    for (int kt = kt0; kt <= kt1; kt += 2) {
        int kb = kt << 4;
        v8f e0, e1;
        {
            v8f c = {};
            c = wmma16(qa0, loadB_f16cg(kbase_p, NHD, lane, kb, seqk, 0, NHD), c);
            c = wmma16(qa1, loadB_f16cg(kbase_p, NHD, lane, kb, seqk, 32, NHD), c);
            e0 = c;
        }
        {
            v8f c = {};
            c = wmma16(qa0, loadB_f16cg(kbase_p, NHD, lane, kb + 16, seqk, 0, NHD), c);
            c = wmma16(qa1, loadB_f16cg(kbase_p, NHD, lane, kb + 16, seqk, 32, NHD), c);
            e1 = c;
        }
        float rowmax[8];
#pragma unroll
        for (int v = 0; v < 8; ++v) rowmax[v] = Mx[v];
#pragma unroll
        for (int s = 0; s < 2; ++s) {
            int kg  = kb + (s << 4) + nl;
            int kgc = kg < seqk ? kg : seqk - 1;
            bool oob = kg >= seqk;
            bool pd = pad[(size_t)b * NS + pad_off + kgc] != 0;
            int segk = (mode == 3) ? seg[(size_t)b * NSL + kgc] : 0;
            v8f& e = s ? e1 : e0;
#pragma unroll
            for (int v = 0; v < 8; ++v) {
                int qg = q0 + hi * 8 + v;
                int relpos, md;
                bool inv = oob | pd;
                if (mode == 0)      { relpos = kg - qg;      md = 12; inv = inv | (relpos > 84) | (relpos < -84); }
                else if (mode == 1) { relpos = kg - segq[v]; md = 4; }
                else if (mode == 2) { relpos = kg - qg;      md = 8; }
                else                { relpos = segk - qg;    md = 4; }
                int relid = relpos;
                if (relid > md) relid = md;
                if (relid < -md) relid = -md;
                relid += md;
                float rv = qrl[(size_t)qg * 32 + relid];
                float en = (e[v] + rv) * INV_SCALE;
                if (inv) en = -1e30f;
                e[v] = en;
                if (en > rowmax[v]) rowmax[v] = en;
            }
        }
#pragma unroll
        for (int v = 0; v < 8; ++v) {
#pragma unroll
            for (int off = 1; off < 16; off <<= 1)
                rowmax[v] = fmaxf(rowmax[v], __shfl_xor(rowmax[v], off, 32));
        }
        float rs[8];
#pragma unroll
        for (int v = 0; v < 8; ++v) {
            float a = __expf(Mx[v] - rowmax[v]);
            Mx[v] = rowmax[v];
            Ls[v] *= a;
            O0[v] *= a; O1[v] *= a; O2[v] *= a;
            float p0 = __expf(e0[v] - Mx[v]);
            float p1 = __expf(e1[v] - Mx[v]);
            rs[v] = p0 + p1;
            int r = hi * 8 + v;
            plds[r * 32 + nl]      = (_Float16)p0;
            plds[r * 32 + 16 + nl] = (_Float16)p1;
        }
#pragma unroll
        for (int v = 0; v < 8; ++v) {
#pragma unroll
            for (int off = 1; off < 16; off <<= 1)
                rs[v] += __shfl_xor(rs[v], off, 32);
            Ls[v] += rs[v];
        }
        asm volatile("s_wait_dscnt 0x0" ::: "memory");
        h8v px = *(const h8v*)&plds[nl * 32 + (hi << 4)];
        h8v py = *(const h8v*)&plds[nl * 32 + 8 + (hi << 4)];
        v16h pa = combine16(px, py);
        asm volatile("s_wait_dscnt 0x0" ::: "memory");
        O0 = wmma16(pa, loadB_vt(vbase_p, seqk, lane, 0, kb, seqk), O0);
        O1 = wmma16(pa, loadB_vt(vbase_p, seqk, lane, 16, kb, seqk), O1);
        O2 = wmma16(pa, loadB_vt(vbase_p, seqk, lane, 32, kb, seqk), O2);
    }
#pragma unroll
    for (int v = 0; v < 8; ++v) {
        float il = 1.0f / fmaxf(Ls[v], 1e-30f);
        int qg = q0 + hi * 8 + v;
        // ctx layout: d = hd*NH + h  (matches transpose(0,2,3,1).reshape)
        size_t base = ((size_t)b * seqq + qg) * NDH + h;
        ctx[base + (size_t)(0  + nl) * NH] = (_Float16)(O0[v] * il);
        ctx[base + (size_t)(16 + nl) * NH] = (_Float16)(O1[v] * il);
        ctx[base + (size_t)(32 + nl) * NH] = (_Float16)(O2[v] * il);
    }
}

// ---------------- Stage 4: output projection z = ctx @ Wo^T -> f32 out slice ----------
__global__ __launch_bounds__(32) void outproj_kernel(const _Float16* __restrict__ ctx,
                                                     const float* __restrict__ Wo,
                                                     float* __restrict__ out,
                                                     int seq, int row_off, int col_off) {
    int lane = threadIdx.x;
    int mt = blockIdx.x, nt = blockIdx.y;
    int rowbase = mt * 16;
    int b = rowbase / seq, q = rowbase % seq;
    const _Float16* Arow = ctx + ((size_t)b * seq + q) * NDH;
    const float* Wrow = Wo + (size_t)nt * 16 * NDH;
    v8f c = {};
    for (int k0 = 0; k0 < NDH; k0 += 32) {
        c = wmma16(loadA_f16(Arow, NDH, lane, k0), loadB_f32(Wrow, NDH, lane, 0, k0), c);
    }
    int n = lane & 15, hi = lane >> 4;
    float* dst = out + ((size_t)b * NS + row_off + q + (hi << 3)) * ND + col_off + nt * 16 + n;
#pragma unroll
    for (int v = 0; v < 8; ++v) dst[(size_t)v * ND] = c[v];
}

// ---------------- Host orchestration ----------------

extern "C" void kernel_launch(void* const* d_in, const int* in_sizes, int n_in,
                              void* d_out, int out_size, void* d_ws, size_t ws_size,
                              hipStream_t stream) {
    (void)in_sizes; (void)n_in; (void)out_size; (void)ws_size;
    const float* query = (const float*)d_in[0];
    const float* key_  = (const float*)d_in[1];
    const float* value = (const float*)d_in[2];
    const unsigned char* pad = (const unsigned char*)d_in[3];
    const int* seg = (const int*)d_in[4];
    const float* Wq_l = (const float*)d_in[5];
    const float* Wq_g = (const float*)d_in[6];
    const float* l2l_Wk = (const float*)d_in[7];
    const float* l2l_Wv = (const float*)d_in[8];
    const float* l2l_Wo = (const float*)d_in[9];
    const float* l2l_rel = (const float*)d_in[10];
    const float* l2g_Wk = (const float*)d_in[11];
    const float* l2g_Wv = (const float*)d_in[12];
    const float* l2g_Wo = (const float*)d_in[13];
    const float* l2g_rel = (const float*)d_in[14];
    const float* g2g_Wk = (const float*)d_in[15];
    const float* g2g_Wv = (const float*)d_in[16];
    const float* g2g_Wo = (const float*)d_in[17];
    const float* g2g_rel = (const float*)d_in[18];
    const float* g2l_Wk = (const float*)d_in[19];
    const float* g2l_Wv = (const float*)d_in[20];
    const float* g2l_Wo = (const float*)d_in[21];
    const float* g2l_rel = (const float*)d_in[22];
    float* out = (float*)d_out;

    char* ws = (char*)d_ws;
    size_t off = 0;
    auto alloc = [&](size_t bytes) -> char* {
        char* p = ws + off;
        off = (off + bytes + 255) & ~(size_t)255;
        return p;
    };
    const size_t fL = (size_t)NB * NH * NSL * NHD; // local proj elements
    const size_t fG = (size_t)NB * NH * NSG * NHD; // global proj elements
    _Float16* qL    = (_Float16*)alloc(fL * 2);
    _Float16* qG    = (_Float16*)alloc(fG * 2);
    _Float16* kL2L  = (_Float16*)alloc(fL * 2);
    _Float16* vTL2L = (_Float16*)alloc(fL * 2);
    _Float16* kL2G  = (_Float16*)alloc(fG * 2);
    _Float16* vTL2G = (_Float16*)alloc(fG * 2);
    _Float16* kG2G  = (_Float16*)alloc(fG * 2);
    _Float16* vTG2G = (_Float16*)alloc(fG * 2);
    _Float16* kG2L  = (_Float16*)alloc(fL * 2);
    _Float16* vTG2L = (_Float16*)alloc(fL * 2);
    float* qrelL2L = (float*)alloc((size_t)NB * NH * NSL * 32 * 4);
    float* qrelL2G = (float*)alloc((size_t)NB * NH * NSL * 32 * 4);
    float* qrelG2G = (float*)alloc((size_t)NB * NH * NSG * 32 * 4);
    float* qrelG2L = (float*)alloc((size_t)NB * NH * NSG * 32 * 4);
    _Float16* ctxL2L = (_Float16*)alloc((size_t)NB * NSL * NDH * 2);
    _Float16* ctxL2G = (_Float16*)alloc((size_t)NB * NSL * NDH * 2);
    _Float16* ctxG2G = (_Float16*)alloc((size_t)NB * NSG * NDH * 2);
    _Float16* ctxG2L = (_Float16*)alloc((size_t)NB * NSG * NDH * 2);

    dim3 blk(32);
    dim3 gpL(NB * NSL / 16, NDH / 16);
    dim3 gpG(NB * NSG / 16, NDH / 16);

    // Stage 1: projections
    proj_kernel<<<gpL, blk, 0, stream>>>(query, Wq_l,   qL,    NSL, 0,   0);
    proj_kernel<<<gpG, blk, 0, stream>>>(query, Wq_g,   qG,    NSG, NSL, 0);
    proj_kernel<<<gpL, blk, 0, stream>>>(key_,  l2l_Wk, kL2L,  NSL, 0,   0);
    proj_kernel<<<gpL, blk, 0, stream>>>(value, l2l_Wv, vTL2L, NSL, 0,   1);
    proj_kernel<<<gpG, blk, 0, stream>>>(key_,  l2g_Wk, kL2G,  NSG, NSL, 0);
    proj_kernel<<<gpG, blk, 0, stream>>>(value, l2g_Wv, vTL2G, NSG, NSL, 1);
    proj_kernel<<<gpG, blk, 0, stream>>>(key_,  g2g_Wk, kG2G,  NSG, NSL, 0);
    proj_kernel<<<gpG, blk, 0, stream>>>(value, g2g_Wv, vTG2G, NSG, NSL, 1);
    proj_kernel<<<gpL, blk, 0, stream>>>(key_,  g2l_Wk, kG2L,  NSL, 0,   0);
    proj_kernel<<<gpL, blk, 0, stream>>>(value, g2l_Wv, vTG2L, NSL, 0,   1);

    // Stage 2: q_rel tables (R padded to 32 columns)
    dim3 grL(NB * NH * NSL / 16, 2);
    dim3 grG(NB * NH * NSG / 16, 2);
    qrel_kernel<<<grL, blk, 0, stream>>>(qL, l2l_rel, qrelL2L, NSL, 2 * 12 + 1);
    qrel_kernel<<<grL, blk, 0, stream>>>(qL, l2g_rel, qrelL2G, NSL, 2 * 4 + 1);
    qrel_kernel<<<grG, blk, 0, stream>>>(qG, g2g_rel, qrelG2G, NSG, 2 * 8 + 1);
    qrel_kernel<<<grG, blk, 0, stream>>>(qG, g2l_rel, qrelG2L, NSG, 2 * 4 + 1);

    // Stage 3: flash attention
    dim3 faL(NSL / 16, NB * NH);
    dim3 faG(NSG / 16, NB * NH);
    flash_kernel<<<faL, blk, 0, stream>>>(qL, kL2L, vTL2L, qrelL2L, pad, seg, ctxL2L, NSL, NSL, 0, 0);
    flash_kernel<<<faL, blk, 0, stream>>>(qL, kL2G, vTL2G, qrelL2G, pad, seg, ctxL2G, NSL, NSG, 1, NSL);
    flash_kernel<<<faG, blk, 0, stream>>>(qG, kG2G, vTG2G, qrelG2G, pad, seg, ctxG2G, NSG, NSG, 2, NSL);
    flash_kernel<<<faG, blk, 0, stream>>>(qG, kG2L, vTG2L, qrelG2L, pad, seg, ctxG2L, NSG, NSL, 3, 0);

    // Stage 4: output projections into concatenated layout
    outproj_kernel<<<gpL, blk, 0, stream>>>(ctxL2L, l2l_Wo, out, NSL, 0,   0);
    outproj_kernel<<<gpL, blk, 0, stream>>>(ctxL2G, l2g_Wo, out, NSL, 0,   NDH);
    outproj_kernel<<<gpG, blk, 0, stream>>>(ctxG2G, g2g_Wo, out, NSG, NSL, 0);
    outproj_kernel<<<gpG, blk, 0, stream>>>(ctxG2L, g2l_Wo, out, NSG, NSL, NDH);
}
